// GDN_60619168416312
// MI455X (gfx1250) — matrix operand order repaired and checked by reading, not compiled, not run on hardware
//
#include <hip/hip_runtime.h>
#include <math.h>

typedef __attribute__((ext_vector_type(16))) _Float16 v16h;
typedef __attribute__((ext_vector_type(8)))  _Float16 v8h;
typedef __attribute__((ext_vector_type(8)))  float    v8f;

constexpr int Bc = 16, Nc = 1000, Tc = 64, DIMc = 64, KTOP = 20;
constexpr int Dc = 128, Hc = 2, DKc = 64, DFFc = 256;
constexpr int BNODES = Bc * Nc;      // 16000
constexpr int ROWS_BT = Bc * Tc;     // 1024

// block tile: 64(M) x 32(N) x 32(K); 4 waves, each wave = 32x16 (2 WMMAs)
#define BM 64
#define BNT 32
#define BKT 32

// ---------------------------------------------------------------------------
// Generic WMMA GEMM: C = act(A[MxK] @ B[KxN] (+bias) (+add))
// transB: B stored [N,K] row-major (used for emb @ emb^T)
// act: 0 none, 1 relu, 2 leaky(0.01), 3 sigmoid
// addMode: 0 none, 1 residual add (addp [M,N]), 2 add addp[(row%addModP)*N+col]
// ---------------------------------------------------------------------------
__global__ __launch_bounds__(128)
void k_gemm(const float* __restrict__ A, const float* __restrict__ Bm,
            const float* __restrict__ bias, const float* __restrict__ addp,
            float* __restrict__ C, int M, int N, int K,
            int transB, int act, int addMode, int addModP)
{
    // row stride 40 halves = 80 bytes (16B aligned) -> b128-friendly, bank-skewed
    __shared__ alignas(16) _Float16 As[BM][BKT + 8];
    __shared__ alignas(16) _Float16 Bt[BNT][BKT + 8];   // transposed: [n][k]

    const int tid  = threadIdx.x;
    const int lane = tid & 31;
    const int wave = tid >> 5;
    const int wr   = wave >> 1;          // 0/1 -> M offset 0/32
    const int wc   = wave & 1;           // 0/1 -> N offset 0/16
    const int half = lane >> 4;
    const int l16  = lane & 15;
    const int m0   = blockIdx.y * BM;
    const int n0   = blockIdx.x * BNT;
    const bool fullM = (m0 + BM <= M);
    const bool fullN = (n0 + BNT <= N);

    v8f acc[2];
    acc[0] = (v8f){0.f,0.f,0.f,0.f,0.f,0.f,0.f,0.f};
    acc[1] = (v8f){0.f,0.f,0.f,0.f,0.f,0.f,0.f,0.f};

    for (int k0 = 0; k0 < K; k0 += BKT) {
        const bool fullK = (k0 + BKT <= K);

        // ---- stage A (f32 -> packed f16), 64x32, pair-packed stores ----
        if (fullM && fullK) {
            for (int i = tid; i < BM * (BKT / 2); i += 128) {
                int r = i >> 4, c = (i & 15) << 1;
                float2 f = *(const float2*)(A + (size_t)(m0 + r) * K + k0 + c);
                union { _Float16 h[2]; unsigned u; } pk;
                pk.h[0] = (_Float16)f.x; pk.h[1] = (_Float16)f.y;
                *(unsigned*)&As[r][c] = pk.u;
            }
        } else {
            for (int i = tid; i < BM * (BKT / 2); i += 128) {
                int r = i >> 4, c = (i & 15) << 1;
                int gr = m0 + r;
                float v0 = 0.f, v1 = 0.f;
                if (gr < M) {
                    if (k0 + c < K)     v0 = A[(size_t)gr * K + k0 + c];
                    if (k0 + c + 1 < K) v1 = A[(size_t)gr * K + k0 + c + 1];
                }
                union { _Float16 h[2]; unsigned u; } pk;
                pk.h[0] = (_Float16)v0; pk.h[1] = (_Float16)v1;
                *(unsigned*)&As[r][c] = pk.u;
            }
        }

        // ---- stage B into transposed LDS tile Bt[n][k] ----
        if (transB) {
            // B stored [N,K]: contiguous along k -> packed loads AND packed stores
            for (int i = tid; i < BNT * (BKT / 2); i += 128) {
                int c = i >> 4, r = (i & 15) << 1;     // c = n, r = k
                int gn = n0 + c;
                float v0 = 0.f, v1 = 0.f;
                if (fullN && fullK) {
                    float2 f = *(const float2*)(Bm + (size_t)gn * K + k0 + r);
                    v0 = f.x; v1 = f.y;
                } else if (gn < N) {
                    if (k0 + r < K)     v0 = Bm[(size_t)gn * K + k0 + r];
                    if (k0 + r + 1 < K) v1 = Bm[(size_t)gn * K + k0 + r + 1];
                }
                union { _Float16 h[2]; unsigned u; } pk;
                pk.h[0] = (_Float16)v0; pk.h[1] = (_Float16)v1;
                *(unsigned*)&Bt[c][r] = pk.u;
            }
        } else {
            // B stored [K,N]: coalesced global read along n, transposed b16 store
            for (int i = tid; i < BKT * BNT; i += 128) {
                int r = i >> 5, c = i & 31;            // r = k, c = n
                int gk = k0 + r, gn = n0 + c;
                float v = (gk < K && gn < N) ? Bm[(size_t)gk * N + gn] : 0.f;
                Bt[c][r] = (_Float16)v;
            }
        }
        __syncthreads();

        // ---- fragment loads: 2x ds_load_b128 per fragment ----
        const int mA = wr * 32 + l16;
        const int nB = wc * 16 + l16;
        v8h a0lo = *(const v8h*)&As[mA][half * 8];
        v8h a0hi = *(const v8h*)&As[mA][16 + half * 8];
        v8h a1lo = *(const v8h*)&As[mA + 16][half * 8];
        v8h a1hi = *(const v8h*)&As[mA + 16][16 + half * 8];
        v8h blo  = *(const v8h*)&Bt[nB][half * 8];
        v8h bhi  = *(const v8h*)&Bt[nB][16 + half * 8];
        v16h af0 = __builtin_shufflevector(a0lo, a0hi, 0,1,2,3,4,5,6,7,8,9,10,11,12,13,14,15);
        v16h af1 = __builtin_shufflevector(a1lo, a1hi, 0,1,2,3,4,5,6,7,8,9,10,11,12,13,14,15);
        v16h bf  = __builtin_shufflevector(blo,  bhi,  0,1,2,3,4,5,6,7,8,9,10,11,12,13,14,15);

        acc[0] = __builtin_amdgcn_wmma_f32_16x16x32_f16(false, af0, false, bf,
                                                        (short)0, acc[0], false, false);
        acc[1] = __builtin_amdgcn_wmma_f32_16x16x32_f16(false, af1, false, bf,
                                                        (short)0, acc[1], false, false);
        __syncthreads();
    }

    // ---- epilogue: bias / residual / PE / activation ----
    int col = n0 + wc * 16 + l16;
    if (col < N) {
#pragma unroll
        for (int p = 0; p < 2; ++p) {
#pragma unroll
            for (int r = 0; r < 8; ++r) {
                int row = m0 + wr * 32 + p * 16 + (half << 3) + r;
                if (row < M) {
                    float v = acc[p][r];
                    if (bias) v += bias[col];
                    if (addMode == 1)      v += addp[(size_t)row * N + col];
                    else if (addMode == 2) v += addp[(size_t)(row % addModP) * N + col];
                    if (act == 1)      v = v > 0.f ? v : 0.f;
                    else if (act == 2) v = v > 0.f ? v : 0.01f * v;
                    else if (act == 3) v = 1.f / (1.f + __expf(-v));
                    C[(size_t)row * N + col] = v;
                }
            }
        }
    }
}

// ---------------------------------------------------------------------------
// Prep: NaN handling, forward/backward series, [B,T,N] X and mask
// ---------------------------------------------------------------------------
__global__ void k_prep(const float* __restrict__ data, float* xf, float* xb,
                       float* Xbt, float* Mbt)
{
    int idx = blockIdx.x * blockDim.x + threadIdx.x;
    int total = Bc * Nc * Tc;
    if (idx >= total) return;
    int t = idx % Tc;
    int n = (idx / Tc) % Nc;
    int b = idx / (Tc * Nc);
    float v = data[idx];
    bool ok = (v == v);
    float x = ok ? v : 0.f;
    float m = ok ? 1.f : 0.f;
    int row = b * Nc + n;
    xf[row * Tc + t] = x;
    xb[row * Tc + (Tc - 1 - t)] = x;
    Xbt[(b * Tc + t) * Nc + n] = x;
    Mbt[(b * Tc + t) * Nc + n] = m;
}

__global__ void k_rnorm(const float* __restrict__ emb, float* rn)
{
    int i = blockIdx.x * blockDim.x + threadIdx.x;
    if (i >= Nc) return;
    float s = 0.f;
    for (int f = 0; f < DIMc; ++f) { float v = emb[i * DIMc + f]; s += v * v; }
    rn[i] = rsqrtf(s);
}

// block per row i: iteratively select top-20 of cos row (ties -> lowest index)
__global__ __launch_bounds__(128)
void k_topk(const float* __restrict__ sim, const float* __restrict__ rn, int* idx)
{
    __shared__ float bv[128];
    __shared__ int   bi[128];
    __shared__ unsigned char sel[Nc];
    int i = blockIdx.x, tid = threadIdx.x;
    for (int j = tid; j < Nc; j += 128) sel[j] = 0;
    __syncthreads();
    float ri = rn[i];
    for (int it = 0; it < KTOP; ++it) {
        float best = -1e30f; int bidx = Nc;
        for (int j = tid; j < Nc; j += 128) {
            if (sel[j]) continue;
            float v = sim[(size_t)i * Nc + j] * ri * rn[j];
            if (v > best || (v == best && j < bidx)) { best = v; bidx = j; }
        }
        bv[tid] = best; bi[tid] = bidx;
        __syncthreads();
        if (tid == 0) {
            float bb = bv[0]; int bj = bi[0];
            for (int t2 = 1; t2 < 128; ++t2)
                if (bv[t2] > bb || (bv[t2] == bb && bi[t2] < bj)) { bb = bv[t2]; bj = bi[t2]; }
            sel[bj] = 1;
            idx[i * KTOP + it] = bj;
        }
        __syncthreads();
    }
}

// per-node attention-projection scalars: pi = g.ai + e.aei ; pj = g.aj + e.aej
__global__ void k_pipj(const float* __restrict__ g, const float* __restrict__ emb,
                       const float* __restrict__ ai, const float* __restrict__ aj,
                       const float* __restrict__ aei, const float* __restrict__ aej,
                       float* pi, float* pj)
{
    int t = blockIdx.x * blockDim.x + threadIdx.x;
    if (t >= BNODES) return;
    int n = t % Nc;
    float s1 = 0.f, s2 = 0.f;
    for (int f = 0; f < DIMc; ++f) {
        float gv = g[t * DIMc + f], ev = emb[n * DIMc + f];
        s1 += gv * ai[f] + ev * aei[f];
        s2 += gv * aj[f] + ev * aej[f];
    }
    pi[t] = s1; pj[t] = s2;
}

// GAT aggregation: each target node has exactly 20 top-k edges + 1 self loop
__global__ __launch_bounds__(64)
void k_gat(const float* __restrict__ g, const float* __restrict__ pi,
           const float* __restrict__ pj, const int* __restrict__ idx,
           const float* __restrict__ bias, float* o)
{
    __shared__ int   ssrc[KTOP + 1];
    __shared__ float sw[KTOP + 1];
    __shared__ unsigned char srm[KTOP + 1];
    int t = blockIdx.x;
    int n = t % Nc, b = t / Nc;
    int tid = threadIdx.x;
    if (tid <= KTOP) {
        int src; int rm = 0;
        if (tid < KTOP) {
            int j = idx[n * KTOP + tid];
            rm = (j == n);
            src = b * Nc + j;
        } else {
            src = t;   // self loop
        }
        float a = pi[t] + pj[src];
        a = a > 0.f ? a : 0.2f * a;       // leaky_relu(0.2)
        if (rm) a = -1e9f;
        ssrc[tid] = src; sw[tid] = a; srm[tid] = (unsigned char)rm;
    }
    __syncthreads();
    if (tid == 0) {
        float m = -1e30f;
        for (int j = 0; j <= KTOP; ++j) m = fmaxf(m, sw[j]);
        float s = 0.f;
        for (int j = 0; j <= KTOP; ++j) {
            float e = srm[j] ? 0.f : __expf(sw[j] - m);
            sw[j] = e; s += e;
        }
        float inv = 1.f / (s + 1e-16f);
        for (int j = 0; j <= KTOP; ++j) sw[j] *= inv;
    }
    __syncthreads();
    float acc = 0.f;
    for (int j = 0; j <= KTOP; ++j) acc += sw[j] * g[ssrc[j] * DIMc + tid];
    o[t * DIMc + tid] = acc + bias[tid];
}

// BatchNorm1d training stats over 16000 rows (block per feature)
__global__ __launch_bounds__(256)
void k_bnstats(const float* __restrict__ o, float* stat)
{
    __shared__ float s1[256], s2[256];
    int f = blockIdx.x, tid = threadIdx.x;
    float a = 0.f, b = 0.f;
    for (int r = tid; r < BNODES; r += 256) {
        float v = o[r * DIMc + f]; a += v; b += v * v;
    }
    s1[tid] = a; s2[tid] = b;
    __syncthreads();
    for (int st = 128; st > 0; st >>= 1) {
        if (tid < st) { s1[tid] += s1[tid + st]; s2[tid] += s2[tid + st]; }
        __syncthreads();
    }
    if (tid == 0) {
        float mu = s1[0] / (float)BNODES;
        float var = s2[0] / (float)BNODES - mu * mu;
        stat[f] = mu; stat[DIMc + f] = var;
    }
}

// BN apply + ReLU + elementwise *emb, write into concat buffer half
__global__ void k_bnapply(const float* __restrict__ o, const float* __restrict__ stat,
                          const float* __restrict__ bng, const float* __restrict__ bnb,
                          const float* __restrict__ emb, float* cat, int off)
{
    int i = blockIdx.x * blockDim.x + threadIdx.x;
    if (i >= BNODES * DIMc) return;
    int f = i % DIMc, t = i / DIMc, n = t % Nc;
    float v = (o[i] - stat[f]) * rsqrtf(stat[DIMc + f] + 1e-5f) * bng[f] + bnb[f];
    v = v > 0.f ? v : 0.f;
    cat[t * (2 * DIMc) + off + f] = v * emb[n * DIMc + f];
}

__global__ void k_cat2(const float* __restrict__ Xbt, const float* __restrict__ Mbt,
                       float* cat2)
{
    int i = blockIdx.x * blockDim.x + threadIdx.x;
    int total = ROWS_BT * 2 * Nc;
    if (i >= total) return;
    int c = i % (2 * Nc), r = i / (2 * Nc);
    cat2[i] = (c < Nc) ? Xbt[r * Nc + c] : Mbt[r * Nc + (c - Nc)];
}

__global__ void k_pe(float* pe)
{
    int i = blockIdx.x * blockDim.x + threadIdx.x;
    if (i >= Tc * Dc) return;
    int d = i % Dc, t = i / Dc;
    float expo = (float)(2 * (d / 2)) / (float)Dc;
    float ang = (float)t * powf(10000.f, -expo);
    pe[i] = (d & 1) ? cosf(ang) : sinf(ang);
}

// LayerNorm over D=128 (block of 128 threads per row)
__global__ __launch_bounds__(128)
void k_ln(const float* __restrict__ x, const float* __restrict__ gm,
          const float* __restrict__ bt, float* y)
{
    __shared__ float s1[128], s2[128];
    int r = blockIdx.x, tid = threadIdx.x;
    float v = x[r * Dc + tid];
    s1[tid] = v; s2[tid] = v * v;
    __syncthreads();
    for (int st = 64; st > 0; st >>= 1) {
        if (tid < st) { s1[tid] += s1[tid + st]; s2[tid] += s2[tid + st]; }
        __syncthreads();
    }
    float mu = s1[0] / (float)Dc;
    float var = s2[0] / (float)Dc - mu * mu;
    y[r * Dc + tid] = (v - mu) * rsqrtf(var + 1e-6f) * gm[tid] + bt[tid];
}

// diag-masked softmax attention for one (b,h); T=DK=64
__global__ __launch_bounds__(64)
void k_attn(const float* __restrict__ q, const float* __restrict__ k,
            const float* __restrict__ v, float* obuf, float* attn)
{
    __shared__ float sk[Tc][DKc];
    __shared__ float sv[Tc][DKc];
    __shared__ float sl[Tc][Tc];
    int bh = blockIdx.x;
    int b = bh / Hc, h = bh % Hc;
    int tid = threadIdx.x;
    for (int i = tid; i < Tc * DKc; i += 64) {
        int s = i / DKc, d = i % DKc;
        sk[s][d] = k[(b * Tc + s) * Dc + h * DKc + d];
        sv[s][d] = v[(b * Tc + s) * Dc + h * DKc + d];
    }
    __syncthreads();
    int t = tid;
    const float* qr = q + (size_t)(b * Tc + t) * Dc + h * DKc;
    float m = -1e30f;
    for (int s = 0; s < Tc; ++s) {
        float a = 0.f;
        for (int d = 0; d < DKc; ++d) a += qr[d] * sk[s][d];
        a *= 0.125f;                // 1/sqrt(64)
        if (s == t) a = -1e9f;      // diagonal mask
        sl[t][s] = a;
        m = fmaxf(m, a);
    }
    float sum = 0.f;
    for (int s = 0; s < Tc; ++s) { float e = __expf(sl[t][s] - m); sl[t][s] = e; sum += e; }
    float inv = 1.f / sum;
    float* arow = attn + (size_t)(bh * Tc + t) * Tc;
    for (int s = 0; s < Tc; ++s) { float w = sl[t][s] * inv; sl[t][s] = w; arow[s] = w; }
    for (int d = 0; d < DKc; ++d) {
        float a = 0.f;
        for (int s = 0; s < Tc; ++s) a += sl[t][s] * sv[s][d];
        obuf[(size_t)(b * Tc + t) * Dc + h * DKc + d] = a;
    }
}

// build [mask, attn.mean(heads)] -> [1024, 1064]
__global__ void k_cwin(const float* __restrict__ Mbt, const float* __restrict__ attn,
                       float* cwin)
{
    int W = Nc + Tc;
    int i = blockIdx.x * blockDim.x + threadIdx.x;
    if (i >= ROWS_BT * W) return;
    int c = i % W, r = i / W;
    if (c < Nc) {
        cwin[i] = Mbt[(size_t)r * Nc + c];
    } else {
        int s = c - Nc; int b = r / Tc, t = r % Tc;
        float a0 = attn[(size_t)((b * Hc + 0) * Tc + t) * Tc + s];
        float a1 = attn[(size_t)((b * Hc + 1) * Tc + t) * Tc + s];
        cwin[i] = 0.5f * (a0 + a1);
    }
}

// combine + write all 6 output tensors (ind_mask == 0 in test mode)
__global__ void k_final(const float* __restrict__ zbuf, const float* __restrict__ enc,
                        const float* __restrict__ cw, const float* __restrict__ Xbt,
                        const float* __restrict__ Mbt, float* out)
{
    int total = ROWS_BT * Nc;
    int i = blockIdx.x * blockDim.x + threadIdx.x;
    if (i >= total) return;
    int n = i % Nc, r = i / Nc;
    int b = r / Tc, t = r % Tc;
    float fu  = zbuf[(size_t)(b * Nc + n) * Tc + t];   // fusion[b,t,n]
    float e   = enc[i];
    float c   = cw[i];
    float cmb = c * e + (1.f - c) * fu;
    float msk = Mbt[i];
    float imp = cmb * (1.f - msk) + Xbt[i] * msk;
    out[i]             = fu;
    out[total + i]     = e;
    out[2 * total + i] = cmb;
    out[3 * total + i] = imp;
    out[4 * total + i] = 0.f;
    out[5 * total + i] = msk;
}

// ---------------------------------------------------------------------------
// Host side
// ---------------------------------------------------------------------------
static inline float* wsf(void* base, size_t& off, size_t count)
{
    float* p = (float*)((char*)base + off);
    off += count * sizeof(float);
    off = (off + 255) & ~(size_t)255;
    return p;
}

static void gemm(hipStream_t st, const float* A, const float* B, const float* bias,
                 const float* addp, float* C, int M, int N, int K,
                 int transB, int act, int addMode, int addModP)
{
    dim3 grid((N + BNT - 1) / BNT, (M + BM - 1) / BM);
    k_gemm<<<grid, 128, 0, st>>>(A, B, bias, addp, C, M, N, K, transB, act, addMode, addModP);
}

extern "C" void kernel_launch(void* const* d_in, const int* in_sizes, int n_in,
                              void* d_out, int out_size, void* d_ws, size_t ws_size,
                              hipStream_t stream)
{
    const float* data  = (const float*)d_in[0];
    const float* embF  = (const float*)d_in[1];
    const float* embB  = (const float*)d_in[2];
    const float* lin[2]  = {(const float*)d_in[3],  (const float*)d_in[11]};
    const float* ai[2]   = {(const float*)d_in[4],  (const float*)d_in[12]};
    const float* aj[2]   = {(const float*)d_in[5],  (const float*)d_in[13]};
    const float* aei[2]  = {(const float*)d_in[6],  (const float*)d_in[14]};
    const float* aej[2]  = {(const float*)d_in[7],  (const float*)d_in[15]};
    const float* gbias[2]= {(const float*)d_in[8],  (const float*)d_in[16]};
    const float* bng[2]  = {(const float*)d_in[9],  (const float*)d_in[17]};
    const float* bnb[2]  = {(const float*)d_in[10], (const float*)d_in[18]};
    const float* emb[2]  = {embF, embB};
    const float* fc_w  = (const float*)d_in[19];
    const float* fc_b  = (const float*)d_in[20];
    const float* e1_w  = (const float*)d_in[21];
    const float* e1_b  = (const float*)d_in[22];
    const float* ln1_g = (const float*)d_in[23];
    const float* ln1_b = (const float*)d_in[24];
    const float* wq    = (const float*)d_in[25];
    const float* wk    = (const float*)d_in[26];
    const float* wv    = (const float*)d_in[27];
    const float* fcp   = (const float*)d_in[28];
    const float* ln2_g = (const float*)d_in[29];
    const float* ln2_b = (const float*)d_in[30];
    const float* w1    = (const float*)d_in[31];
    const float* b1    = (const float*)d_in[32];
    const float* w2    = (const float*)d_in[33];
    const float* b2    = (const float*)d_in[34];
    const float* red_w = (const float*)d_in[35];
    const float* red_b = (const float*)d_in[36];
    const float* wc_w  = (const float*)d_in[37];
    const float* wc_b  = (const float*)d_in[38];
    (void)in_sizes; (void)n_in; (void)out_size; (void)ws_size;

    // ---- workspace carve ----
    size_t off = 0;
    float* sim   = wsf(d_ws, off, (size_t)Nc * Nc);
    float* rn    = wsf(d_ws, off, 1024);
    int*   idxF  = (int*)wsf(d_ws, off, Nc * KTOP);
    int*   idxB  = (int*)wsf(d_ws, off, Nc * KTOP);
    int*   idxD[2] = {idxF, idxB};
    float* xf    = wsf(d_ws, off, (size_t)BNODES * Tc);
    float* xb    = wsf(d_ws, off, (size_t)BNODES * Tc);
    const float* xdir[2] = {xf, xb};
    float* Xbt   = wsf(d_ws, off, (size_t)ROWS_BT * Nc);
    float* Mbt   = wsf(d_ws, off, (size_t)ROWS_BT * Nc);
    float* gbuf  = wsf(d_ws, off, (size_t)BNODES * DIMc);
    float* pib   = wsf(d_ws, off, BNODES);
    float* pjb   = wsf(d_ws, off, BNODES);
    float* ogat  = wsf(d_ws, off, (size_t)BNODES * DIMc);
    float* stat  = wsf(d_ws, off, 2 * DIMc);
    float* catb  = wsf(d_ws, off, (size_t)BNODES * 2 * DIMc);
    float* zbuf  = wsf(d_ws, off, (size_t)BNODES * Tc);
    float* cat2  = wsf(d_ws, off, (size_t)ROWS_BT * 2 * Nc);
    float* pe    = wsf(d_ws, off, Tc * Dc);
    float* hA    = wsf(d_ws, off, ROWS_BT * Dc);
    float* hB    = wsf(d_ws, off, ROWS_BT * Dc);
    float* ybuf  = wsf(d_ws, off, ROWS_BT * Dc);
    float* qb    = wsf(d_ws, off, ROWS_BT * Dc);
    float* kb    = wsf(d_ws, off, ROWS_BT * Dc);
    float* vb    = wsf(d_ws, off, ROWS_BT * Dc);
    float* obuf  = wsf(d_ws, off, ROWS_BT * Dc);
    float* ff1   = wsf(d_ws, off, ROWS_BT * DFFc);
    float* attn  = wsf(d_ws, off, (size_t)Bc * Hc * Tc * Tc);
    float* encb  = wsf(d_ws, off, (size_t)ROWS_BT * Nc);
    float* cwin  = wsf(d_ws, off, (size_t)ROWS_BT * (Nc + Tc));
    float* cwb   = wsf(d_ws, off, (size_t)ROWS_BT * Nc);

    // ---- 1. prep ----
    {
        int total = Bc * Nc * Tc;
        k_prep<<<(total + 255) / 256, 256, 0, stream>>>(data, xf, xb, Xbt, Mbt);
    }

    // ---- 2. graph construction (top-k cosine) per direction ----
    for (int dir = 0; dir < 2; ++dir) {
        k_rnorm<<<(Nc + 127) / 128, 128, 0, stream>>>(emb[dir], rn);
        gemm(stream, emb[dir], emb[dir], nullptr, nullptr, sim,
             Nc, Nc, DIMc, /*transB=*/1, 0, 0, 0);
        k_topk<<<Nc, 128, 0, stream>>>(sim, rn, idxD[dir]);
    }

    // ---- 3. GAT per direction -> concat buffer ----
    for (int dir = 0; dir < 2; ++dir) {
        gemm(stream, xdir[dir], lin[dir], nullptr, nullptr, gbuf,
             BNODES, DIMc, Tc, 0, 0, 0, 0);
        k_pipj<<<(BNODES + 255) / 256, 256, 0, stream>>>(
            gbuf, emb[dir], ai[dir], aj[dir], aei[dir], aej[dir], pib, pjb);
        k_gat<<<BNODES, 64, 0, stream>>>(gbuf, pib, pjb, idxD[dir], gbias[dir], ogat);
        k_bnstats<<<DIMc, 256, 0, stream>>>(ogat, stat);
        k_bnapply<<<(BNODES * DIMc + 255) / 256, 256, 0, stream>>>(
            ogat, stat, bng[dir], bnb[dir], emb[dir], catb, dir * DIMc);
    }

    // ---- 4. fusion = leaky_0.01(cat @ fc_w + fc_b), stored [B,N,T] ----
    gemm(stream, catb, fc_w, fc_b, nullptr, zbuf, BNODES, Tc, 2 * DIMc, 0, 2, 0, 0);

    // ---- 5. SAITS embedding + positional encoding ----
    k_cat2<<<(ROWS_BT * 2 * Nc + 255) / 256, 256, 0, stream>>>(Xbt, Mbt, cat2);
    k_pe<<<(Tc * Dc + 255) / 256, 256, 0, stream>>>(pe);
    gemm(stream, cat2, e1_w, e1_b, pe, hA, ROWS_BT, Dc, 2 * Nc, 0, 0, /*addMode=*/2, Tc);

    // ---- 6. 9 transformer blocks (L outer x L inner, layer params l = it/3) ----
    for (int it = 0; it < 9; ++it) {
        int l = it / 3;
        const float* wq_l  = wq  + (size_t)l * Dc * Dc;
        const float* wk_l  = wk  + (size_t)l * Dc * Dc;
        const float* wv_l  = wv  + (size_t)l * Dc * Dc;
        const float* fcp_l = fcp + (size_t)l * Dc * Dc;
        const float* w1_l  = w1  + (size_t)l * Dc * DFFc;
        const float* b1_l  = b1  + (size_t)l * DFFc;
        const float* w2_l  = w2  + (size_t)l * DFFc * Dc;
        const float* b2_l  = b2  + (size_t)l * Dc;

        k_ln<<<ROWS_BT, Dc, 0, stream>>>(hA, ln1_g + l * Dc, ln1_b + l * Dc, ybuf);
        gemm(stream, ybuf, wq_l, nullptr, nullptr, qb, ROWS_BT, Dc, Dc, 0, 0, 0, 0);
        gemm(stream, ybuf, wk_l, nullptr, nullptr, kb, ROWS_BT, Dc, Dc, 0, 0, 0, 0);
        gemm(stream, ybuf, wv_l, nullptr, nullptr, vb, ROWS_BT, Dc, Dc, 0, 0, 0, 0);
        k_attn<<<Bc * Hc, Tc, 0, stream>>>(qb, kb, vb, obuf, attn);
        gemm(stream, obuf, fcp_l, nullptr, hA, hB, ROWS_BT, Dc, Dc, 0, 0, /*res*/1, 0);
        k_ln<<<ROWS_BT, Dc, 0, stream>>>(hB, ln2_g + l * Dc, ln2_b + l * Dc, ybuf);
        gemm(stream, ybuf, w1_l, b1_l, nullptr, ff1, ROWS_BT, DFFc, Dc, 0, /*relu*/1, 0, 0);
        gemm(stream, ff1, w2_l, b2_l, hB, hA, ROWS_BT, Dc, DFFc, 0, 0, /*res*/1, 0);
    }

    // ---- 7. reduce to [B,T,N], combining weights, final outputs ----
    gemm(stream, hA, red_w, red_b, nullptr, encb, ROWS_BT, Nc, Dc, 0, 0, 0, 0);
    k_cwin<<<(ROWS_BT * (Nc + Tc) + 255) / 256, 256, 0, stream>>>(Mbt, attn, cwin);
    gemm(stream, cwin, wc_w, wc_b, nullptr, cwb, ROWS_BT, Nc, Nc + Tc, 0, /*sigmoid*/3, 0, 0);
    k_final<<<(ROWS_BT * Nc + 255) / 256, 256, 0, stream>>>(
        zbuf, encb, cwb, Xbt, Mbt, (float*)d_out);
}
